// Attention_83657372992139
// MI455X (gfx1250) — matrix operand order repaired and checked by reading, not compiled, not run on hardware
//
#include <hip/hip_runtime.h>

typedef __attribute__((ext_vector_type(16))) _Float16 v16h;
typedef __attribute__((ext_vector_type(8)))  _Float16 v8h;
typedef __attribute__((ext_vector_type(4)))  _Float16 v4h;
typedef __attribute__((ext_vector_type(8)))  float    v8f;
typedef __attribute__((ext_vector_type(4)))  float    v4f;

#define ATT_B 4
#define ATT_H 16
#define ATT_S 2048
#define ATT_D 64

#define QROWS 16              // q rows per wave
#define WAVES 8               // waves per block
#define QBLK  (QROWS * WAVES) // 128 q rows per block
#define PROW  72              // LDS row stride in halfs for 64-wide P (conflict-free, 16B-aligned)

// 1/sqrt(D) * log2(e): scores land in base-2 domain -> raw v_exp_f32, no mul
#define QSCALE (0.125f * 1.4426950408889634f)

// D = A*B + C, A/B f16 16x32 / 32x16, C/D f32 16x16
__device__ __forceinline__ v8f wmma_f16(v16h a, v16h b, v8f c) {
  return __builtin_amdgcn_wmma_f32_16x16x32_f16(false, a, false, b, (short)0, c,
                                                false, false);
}

// 16 contiguous halfs (16B-aligned) -> fragment slots 0..15
__device__ __forceinline__ v16h ldh16(const _Float16* p) {
  v8h a = *(const v8h*)p;
  v8h b = *(const v8h*)(p + 8);
  return __builtin_shufflevector(a, b, 0, 1, 2, 3, 4, 5, 6, 7,
                                 8, 9, 10, 11, 12, 13, 14, 15);
}

// A-fragment gap load: slots 0..7 = p[0..8), slots 8..15 = p[16..24)
__device__ __forceinline__ v16h ldh16_gap(const _Float16* p) {
  v8h a = *(const v8h*)p;
  v8h b = *(const v8h*)(p + 16);
  return __builtin_shufflevector(a, b, 0, 1, 2, 3, 4, 5, 6, 7,
                                 8, 9, 10, 11, 12, 13, 14, 15);
}

// convert 8 contiguous f32 (16B-aligned) into fragment slots [base, base+8)
__device__ __forceinline__ void cvt8(v16h& r, int base, const float* p, float s) {
  const v4f* q = (const v4f*)p;
  v4f x0 = q[0], x1 = q[1];
  r[base + 0] = (_Float16)(x0[0] * s);
  r[base + 1] = (_Float16)(x0[1] * s);
  r[base + 2] = (_Float16)(x0[2] * s);
  r[base + 3] = (_Float16)(x0[3] * s);
  r[base + 4] = (_Float16)(x1[0] * s);
  r[base + 5] = (_Float16)(x1[1] * s);
  r[base + 6] = (_Float16)(x1[2] * s);
  r[base + 7] = (_Float16)(x1[3] * s);
}

// ---------------------------------------------------------------------------
// Pre-pass: K -> f16 row-major, V -> f16 transposed Vt[bh][d][s]
// ---------------------------------------------------------------------------
__global__ __launch_bounds__(256) void
cvt_kv(const float* __restrict__ K, const float* __restrict__ V,
       _Float16* __restrict__ K16, _Float16* __restrict__ Vt)
{
  const size_t total = (size_t)ATT_B * ATT_H * ATT_S * ATT_D;
  const size_t i4 = ((size_t)blockIdx.x * blockDim.x + threadIdx.x) * 4;
  if (i4 >= total) return;

  v4f kx = *(const v4f*)(K + i4);
  v4h kh;
  kh[0] = (_Float16)kx[0]; kh[1] = (_Float16)kx[1];
  kh[2] = (_Float16)kx[2]; kh[3] = (_Float16)kx[3];
  *(v4h*)(K16 + i4) = kh;

  v4f vx = *(const v4f*)(V + i4);
  const int    d    = (int)(i4 & (ATT_D - 1));
  const size_t srow = i4 / ATT_D;                  // bh*S + s
  const int    s    = (int)(srow & (ATT_S - 1));
  const size_t bh   = srow / ATT_S;
  _Float16* vt = Vt + ((bh * ATT_D + d) * (size_t)ATT_S + s);
#pragma unroll
  for (int t = 0; t < 4; ++t) vt[(size_t)t * ATT_S] = (_Float16)vx[t];
}

// Pre-pass: attention_mask -> additive bias (0 / -inf)
__global__ __launch_bounds__(256) void
cvt_bias(const int* __restrict__ mask, float* __restrict__ bias)
{
  const int i = blockIdx.x * blockDim.x + threadIdx.x;
  if (i < ATT_B * ATT_S) bias[i] = (mask[i] != 0) ? 0.0f : -__builtin_inff();
}

// ---------------------------------------------------------------------------
// One 64-key chunk: QK^T, online softmax (base-2), P@V.
// ---------------------------------------------------------------------------
template <bool CAUSAL>
__device__ __forceinline__ void
fa_chunk(int cb, const _Float16* Kb, const _Float16* Vtb,
         const float* biasb, _Float16* pw,
         int col, int hi, int kb, int qrow0,
         const v16h& aq0, const v16h& aq1, const v16h& vones,
         v8f* acc, v8f& racc, float* rmax)
{
  // ---- S = Q @ K^T : 4 N-tiles of 16 keys ----
  v8f sc[4];
#pragma unroll
  for (int nt = 0; nt < 4; ++nt) {
    const _Float16* kp = Kb + (size_t)(cb + nt * 16 + col) * ATT_D + hi * 8;
    v16h klo = ldh16_gap(kp);
    v16h khi = ldh16_gap(kp + 32);
    v8f s = {};
    s = wmma_f16(aq0, klo, s);
    s = wmma_f16(aq1, khi, s);
    sc[nt] = s;
  }
  __builtin_prefetch(Kb + (size_t)(cb + 64 + col) * ATT_D, 0, 3);

  // padding bias (0 / -inf), vector add -> v_pk_add_f32
  v8f xv[4];
#pragma unroll
  for (int nt = 0; nt < 4; ++nt) {
    xv[nt] = sc[nt] + biasb[cb + nt * 16 + col];
    if (CAUSAL) {
      const int kv = cb + nt * 16 + col;
#pragma unroll
      for (int j = 0; j < 8; ++j)
        xv[nt][j] = (kv <= qrow0 + j) ? xv[nt][j] : -__builtin_inff();
    }
  }

  // ---- online softmax (base 2): one rescale per 64 keys ----
  v8f corrv;
#pragma unroll
  for (int j = 0; j < 8; ++j) {
    float cm = fmaxf(fmaxf(xv[0][j], xv[1][j]), fmaxf(xv[2][j], xv[3][j]));
    cm = fmaxf(cm, __shfl_xor(cm, 1));
    cm = fmaxf(cm, __shfl_xor(cm, 2));
    cm = fmaxf(cm, __shfl_xor(cm, 4));
    cm = fmaxf(cm, __shfl_xor(cm, 8));
    const float nm = fmaxf(rmax[j], cm);
    corrv[j] = __builtin_amdgcn_exp2f(rmax[j] - nm);
    rmax[j] = nm;
#pragma unroll
    for (int nt = 0; nt < 4; ++nt) {
      const float p = __builtin_amdgcn_exp2f(xv[nt][j] - nm);
      pw[(kb + j) * PROW + nt * 16 + col] = (_Float16)p;
    }
  }
  // vector rescale -> v_pk_mul_f32
  acc[0] *= corrv; acc[1] *= corrv; acc[2] *= corrv; acc[3] *= corrv;
  racc   *= corrv;

  __builtin_amdgcn_wave_barrier();  // DS pipe in-order per wave; stop reordering

  // ---- reload P as two A-fragments (k = 0..31, 32..63) ----
  const _Float16* pr = pw + col * PROW + kb;
  v16h ap0 = ldh16_gap(pr);
  v16h ap1 = ldh16_gap(pr + 32);

  __builtin_amdgcn_wave_barrier();

  // ---- O += P @ V (Vt pre-transposed: contiguous along s) ----
#pragma unroll
  for (int dsl = 0; dsl < 4; ++dsl) {
    const _Float16* vp = Vtb + (size_t)(dsl * 16 + col) * ATT_S + cb + hi * 8;
    v16h v0 = ldh16_gap(vp);
    v16h v1 = ldh16_gap(vp + 32);
    acc[dsl] = wmma_f16(ap0, v0, acc[dsl]);
    acc[dsl] = wmma_f16(ap1, v1, acc[dsl]);
  }
  racc = wmma_f16(ap0, vones, racc);
  racc = wmma_f16(ap1, vones, racc);
}

// ---------------------------------------------------------------------------
// Main flash-attention kernel using pre-converted K16 / Vt / bias
// ---------------------------------------------------------------------------
__global__ __launch_bounds__(QBLK * 2) void
fa_fwd_ws(const float* __restrict__ Q, const _Float16* __restrict__ K16,
          const _Float16* __restrict__ Vt, const float* __restrict__ bias,
          float* __restrict__ O)
{
  __shared__ __align__(16) _Float16 pbuf[WAVES * QROWS * PROW];

  const int tid  = threadIdx.x;
  const int lane = tid & 31;
  const int wave = tid >> 5;
  const int col  = lane & 15;
  const int hi   = lane >> 4;

  const int tiles = ATT_S / QBLK;
  const int bh    = blockIdx.x / tiles;
  const int tile  = blockIdx.x % tiles;
  const int b     = bh / ATT_H;
  const int qw    = tile * QBLK + wave * QROWS;

  const size_t bhS = (size_t)bh * ATT_S;
  const float*    Qb    = Q   + bhS * ATT_D;
  const _Float16* Kb    = K16 + bhS * ATT_D;
  const _Float16* Vtb   = Vt  + (size_t)bh * ATT_D * ATT_S;
  float*          Ob    = O   + bhS * ATT_D;
  const float*    biasb = bias + (size_t)b * ATT_S;

  _Float16* pw = pbuf + wave * (QROWS * PROW);
  const int kb = hi * 8;

  // ---- Q A-fragments (row = qw+col), 1/sqrt(64)*log2(e) folded in ----
  const float* qrow = Qb + (size_t)(qw + col) * ATT_D;
  v16h aq0, aq1;
  cvt8(aq0, 0, qrow + kb,           QSCALE);
  cvt8(aq0, 8, qrow + kb + 16,      QSCALE);
  cvt8(aq1, 0, qrow + 32 + kb,      QSCALE);
  cvt8(aq1, 8, qrow + 32 + kb + 16, QSCALE);

  v16h vones;
#pragma unroll
  for (int j = 0; j < 16; ++j) vones[j] = (_Float16)1.0f;

  v8f acc[4] = {{}, {}, {}, {}};
  v8f racc = {};
  float rmax[8];
#pragma unroll
  for (int j = 0; j < 8; ++j) rmax[j] = -3.0e38f;

  const int qrow0 = qw + hi * 8;
  const int nfull = qw >> 6;                  // chunks fully below the diagonal
  const int nch   = (qw + QROWS + 63) >> 6;   // total 64-key chunks (causal bound)

  for (int c = 0; c < nfull; ++c)
    fa_chunk<false>(c << 6, Kb, Vtb, biasb, pw, col, hi, kb, qrow0,
                    aq0, aq1, vones, acc, racc, rmax);
  for (int c = nfull; c < nch; ++c)
    fa_chunk<true>(c << 6, Kb, Vtb, biasb, pw, col, hi, kb, qrow0,
                   aq0, aq1, vones, acc, racc, rmax);

  // ---- epilogue: each lane's slots already hold their rows' sums ----
#pragma unroll
  for (int j = 0; j < 8; ++j) {
    const float inv = 1.0f / racc[j];
    float* op = Ob + (size_t)(qrow0 + j) * ATT_D + col;
    op[0]  = acc[0][j] * inv;
    op[16] = acc[1][j] * inv;
    op[32] = acc[2][j] * inv;
    op[48] = acc[3][j] * inv;
  }
}

// ---------------------------------------------------------------------------
// Fallback (no workspace): self-contained kernel
// ---------------------------------------------------------------------------
#define PROWF 40
__global__ __launch_bounds__(QBLK * 2) void
fa_fwd_nows(const float* __restrict__ Q, const float* __restrict__ K,
            const float* __restrict__ V, const int* __restrict__ mask,
            float* __restrict__ O)
{
  __shared__ __align__(16) _Float16 pbuf[WAVES * QROWS * PROWF];

  const int tid  = threadIdx.x;
  const int lane = tid & 31;
  const int wave = tid >> 5;
  const int col  = lane & 15;
  const int hi   = lane >> 4;

  const int tiles = ATT_S / QBLK;
  const int bh    = blockIdx.x / tiles;
  const int tile  = blockIdx.x % tiles;
  const int b     = bh / ATT_H;
  const int qw    = tile * QBLK + wave * QROWS;

  const size_t bhS = (size_t)bh * ATT_S;
  const float* Qb = Q + bhS * ATT_D;
  const float* Kb = K + bhS * ATT_D;
  const float* Vb = V + bhS * ATT_D;
  float*       Ob = O + bhS * ATT_D;
  const int*   mb = mask + (size_t)b * ATT_S;

  _Float16* pw = pbuf + wave * (QROWS * PROWF);
  const int kb = hi * 8;

  const float* qrow = Qb + (size_t)(qw + col) * ATT_D;
  v16h aq0, aq1;
  cvt8(aq0, 0, qrow + kb,           QSCALE);
  cvt8(aq0, 8, qrow + kb + 16,      QSCALE);
  cvt8(aq1, 0, qrow + 32 + kb,      QSCALE);
  cvt8(aq1, 8, qrow + 32 + kb + 16, QSCALE);

  v8f acc0 = {}, acc1 = {}, acc2 = {}, acc3 = {};
  float rmax[8], rsum[8];
#pragma unroll
  for (int j = 0; j < 8; ++j) { rmax[j] = -3.0e38f; rsum[j] = 0.0f; }

  const int qrow0 = qw + hi * 8;
  const int nch   = (qw + QROWS + 31) >> 5;

  for (int c = 0; c < nch; ++c) {
    const int cb = c << 5;
    const float* kp1 = Kb + (size_t)(cb + col) * ATT_D + hi * 8;
    const float* kp2 = kp1 + (size_t)16 * ATT_D;
    v16h b00, b01, b10, b11;
    cvt8(b00, 0, kp1,      1.f); cvt8(b00, 8, kp1 + 16, 1.f);
    cvt8(b01, 0, kp1 + 32, 1.f); cvt8(b01, 8, kp1 + 48, 1.f);
    cvt8(b10, 0, kp2,      1.f); cvt8(b10, 8, kp2 + 16, 1.f);
    cvt8(b11, 0, kp2 + 32, 1.f); cvt8(b11, 8, kp2 + 48, 1.f);

    v8f s1 = {}, s2 = {};
    s1 = wmma_f16(aq0, b00, s1);
    s1 = wmma_f16(aq1, b01, s1);
    s2 = wmma_f16(aq0, b10, s2);
    s2 = wmma_f16(aq1, b11, s2);

    const int kv1 = cb + col, kv2 = kv1 + 16;
    const bool ok1 = (mb[kv1] != 0);
    const bool ok2 = (mb[kv2] != 0);

#pragma unroll
    for (int j = 0; j < 8; ++j) {
      const int qr = qrow0 + j;
      float x1 = (ok1 && kv1 <= qr) ? s1[j] : -__builtin_inff();
      float x2 = (ok2 && kv2 <= qr) ? s2[j] : -__builtin_inff();
      float cm = fmaxf(x1, x2);
      cm = fmaxf(cm, __shfl_xor(cm, 1));
      cm = fmaxf(cm, __shfl_xor(cm, 2));
      cm = fmaxf(cm, __shfl_xor(cm, 4));
      cm = fmaxf(cm, __shfl_xor(cm, 8));
      const float nm   = fmaxf(rmax[j], cm);
      const float corr = __builtin_amdgcn_exp2f(rmax[j] - nm);
      rmax[j] = nm;
      const float p1 = __builtin_amdgcn_exp2f(x1 - nm);
      const float p2 = __builtin_amdgcn_exp2f(x2 - nm);
      float ps = p1 + p2;
      ps += __shfl_xor(ps, 1);
      ps += __shfl_xor(ps, 2);
      ps += __shfl_xor(ps, 4);
      ps += __shfl_xor(ps, 8);
      rsum[j] = rsum[j] * corr + ps;
      acc0[j] *= corr; acc1[j] *= corr; acc2[j] *= corr; acc3[j] *= corr;
      pw[(kb + j) * PROWF + col]      = (_Float16)p1;
      pw[(kb + j) * PROWF + 16 + col] = (_Float16)p2;
    }

    __builtin_amdgcn_wave_barrier();
    const _Float16* pr = pw + col * PROWF + kb;
    v16h ap = ldh16_gap(pr);
    __builtin_amdgcn_wave_barrier();

    const float* vp = Vb + (size_t)(cb + hi * 8) * ATT_D + col;
    v16h bv0, bv1, bv2, bv3;
#pragma unroll
    for (int j = 0; j < 16; ++j) {
      const float* r = vp + (size_t)(j < 8 ? j : j + 8) * ATT_D;
      bv0[j] = (_Float16)r[0];
      bv1[j] = (_Float16)r[16];
      bv2[j] = (_Float16)r[32];
      bv3[j] = (_Float16)r[48];
    }
    acc0 = wmma_f16(ap, bv0, acc0);
    acc1 = wmma_f16(ap, bv1, acc1);
    acc2 = wmma_f16(ap, bv2, acc2);
    acc3 = wmma_f16(ap, bv3, acc3);
  }

#pragma unroll
  for (int j = 0; j < 8; ++j) {
    const float inv = 1.0f / rsum[j];
    float* op = Ob + (size_t)(qrow0 + j) * ATT_D + col;
    op[0]  = acc0[j] * inv;
    op[16] = acc1[j] * inv;
    op[32] = acc2[j] * inv;
    op[48] = acc3[j] * inv;
  }
}

extern "C" void kernel_launch(void* const* d_in, const int* in_sizes, int n_in,
                              void* d_out, int out_size, void* d_ws, size_t ws_size,
                              hipStream_t stream) {
  (void)in_sizes; (void)n_in; (void)out_size;
  const float* q = (const float*)d_in[0];
  const float* k = (const float*)d_in[1];
  const float* v = (const float*)d_in[2];
  const int*   m = (const int*)d_in[3];
  float*       o = (float*)d_out;

  const int blocks = ATT_B * ATT_H * (ATT_S / QBLK);  // 1024 blocks, 8 waves each
  const size_t nelem = (size_t)ATT_B * ATT_H * ATT_S * ATT_D;
  const size_t need  = 2 * nelem * sizeof(_Float16) + (size_t)ATT_B * ATT_S * sizeof(float);

  if (ws_size >= need) {
    _Float16* K16  = (_Float16*)d_ws;
    _Float16* Vt   = K16 + nelem;
    float*    bias = (float*)(Vt + nelem);
    const int cblocks = (int)((nelem / 4 + 255) / 256);
    cvt_kv<<<cblocks, 256, 0, stream>>>(k, v, K16, Vt);
    cvt_bias<<<(ATT_B * ATT_S + 255) / 256, 256, 0, stream>>>(m, bias);
    fa_fwd_ws<<<blocks, QBLK * 2, 0, stream>>>(q, K16, Vt, bias, o);
  } else {
    fa_fwd_nows<<<blocks, QBLK * 2, 0, stream>>>(q, k, v, m, o);
  }
}
